// MultiHeadAttention_21388937134351
// MI455X (gfx1250) — matrix operand order repaired and checked by reading, not compiled, not run on hardware
//
#include <hip/hip_runtime.h>
#include <hip/hip_bf16.h>

// ---------------- problem constants ----------------
constexpr int Bc  = 4;
constexpr int Sc  = 2048;
constexpr int DMc = 512;
constexpr int Hc  = 8;
constexpr int Dc  = 64;                 // depth per head
constexpr float SCALE = 0.125f;         // 1/sqrt(64)

typedef __attribute__((ext_vector_type(16))) __bf16 v16bf;
typedef __attribute__((ext_vector_type(8)))  float  v8f;
typedef int v4i __attribute__((vector_size(16)));   // matches builtin param type

#define ATHENA_ASYNC __has_builtin(__builtin_amdgcn_global_load_async_to_lds_b128)

__device__ inline v8f wmma_bf16(v16bf a, v16bf b, v8f c) {
  return __builtin_amdgcn_wmma_f32_16x16x32_bf16(
      false, a, false, b, (short)0, c, false, false);
}

// Fragment loader: two contiguous 16B runs per lane (K = kb..kb+7, kb+16..kb+23)
// matches the CDNA5 16-bit A/B VGPR layout (ISA 7.12.2). base must be 16B aligned.
__device__ inline v16bf frag_2x16B(const __bf16* base, int kb) {
  union { uint4 u[2]; v16bf v; } t;
  t.u[0] = *(const uint4*)(base + kb);
  t.u[1] = *(const uint4*)(base + 16 + kb);
  return t.v;
}

// 16-byte global -> LDS copy: async on CDNA5 (GLOBAL_LOAD_ASYNC_TO_LDS_B128),
// plain vector copy fallback.
__device__ inline void cp16_g2l(void* lds_dst, const void* gsrc) {
#if ATHENA_ASYNC
  __builtin_amdgcn_global_load_async_to_lds_b128(
      (__attribute__((address_space(1))) v4i*)gsrc,
      (__attribute__((address_space(3))) v4i*)lds_dst, 0, 0);
#else
  *(uint4*)lds_dst = *(const uint4*)gsrc;
#endif
}
__device__ inline void async_join() {
#if ATHENA_ASYNC
#if __has_builtin(__builtin_amdgcn_s_wait_asynccnt)
  __builtin_amdgcn_s_wait_asynccnt(0);
#else
  asm volatile("s_wait_asynccnt 0" ::: "memory");
#endif
#endif
}

// ======================================================================
// GEMM: C[M=8192,512] = X[8192,512] * W[512,512] + bias
// MODE 0: out f32 row-major [M,512]
// MODE 1: out bf16 head-split [B,H,S,64]
// MODE 2: out bf16 head-split transposed [B,H,64,S]   (for V)
// TX: float (stage converts) or __bf16 (raw async copy)
// ======================================================================
template <int MODE, typename TX>
__global__ __launch_bounds__(256)
void gemm512(const TX* __restrict__ X, const float* __restrict__ W,
             const float* __restrict__ bias, void* __restrict__ outp) {
  __shared__ alignas(16) __bf16 xa[128][32];        // A tile, K-contiguous rows
  __shared__ alignas(16) __bf16 wbT[64][32 + 8];    // B tile TRANSPOSED [n][k]

  const int m0   = blockIdx.x * 128;
  const int n0   = blockIdx.y * 64;
  const int tid  = threadIdx.x;
  const int lane = tid & 31;
  const int wv   = tid >> 5;

  v8f acc[4] = {};
  const int kb = (lane < 16) ? 0 : 8;

  for (int k0 = 0; k0 < DMc; k0 += 32) {
    __syncthreads();
    if constexpr (sizeof(TX) == 2) {
      // bf16 input: 512 x 16B chunks, async copy
      for (int idx = tid; idx < 512; idx += 256) {
        int r = idx >> 2, q = idx & 3;
        cp16_g2l(&xa[r][q * 8],
                 (const __bf16*)X + (size_t)(m0 + r) * DMc + k0 + q * 8);
      }
    } else {
      // f32 input: vector load + convert + packed 8B store
      for (int idx = tid; idx < 1024; idx += 256) {
        int r = idx >> 3, c = (idx & 7) * 4;
        float4 f = *(const float4*)((const float*)X + (size_t)(m0 + r) * DMc + k0 + c);
        union { uint2 u; __bf16 h[4]; } p;
        p.h[0] = (__bf16)f.x; p.h[1] = (__bf16)f.y;
        p.h[2] = (__bf16)f.z; p.h[3] = (__bf16)f.w;
        *(uint2*)&xa[r][c] = p.u;
      }
    }
    // W tile: coalesced f32 read, transposed bf16 store
    for (int i = tid; i < 32 * 64; i += 256) {
      int r = i >> 6, c = i & 63;
      wbT[c][r] = (__bf16)W[(size_t)(k0 + r) * DMc + (n0 + c)];
    }
    async_join();
    __syncthreads();

    const v16bf a = frag_2x16B(&xa[wv * 16 + (lane & 15)][0], kb);
#pragma unroll
    for (int f = 0; f < 4; ++f) {
      const v16bf bfrag = frag_2x16B(&wbT[f * 16 + (lane & 15)][0], kb);
      acc[f] = wmma_bf16(a, bfrag, acc[f]);
    }
  }

  const int n  = lane & 15;
  const int hi = lane >> 4;
#pragma unroll
  for (int f = 0; f < 4; ++f) {
    const int c = n0 + f * 16 + n;
    const float bv = bias[c];
#pragma unroll
    for (int r = 0; r < 8; ++r) {
      const int m = m0 + wv * 16 + r + hi * 8;      // token 0..8191
      const float vout = acc[f][r] + bv;
      if constexpr (MODE == 0) {
        ((float*)outp)[(size_t)m * DMc + c] = vout;
      } else {
        const int b = m >> 11, s = m & (Sc - 1);
        const int h = c >> 6, d = c & 63;
        if constexpr (MODE == 1)
          ((__bf16*)outp)[(((size_t)b * Hc + h) * Sc + s) * Dc + d] = (__bf16)vout;
        else
          ((__bf16*)outp)[(((size_t)b * Hc + h) * Dc + d) * Sc + s] = (__bf16)vout;
      }
    }
  }
}

// ======================================================================
// Attention: block = (b, h, 128-query tile), 8 waves x 16 rows.
// Pass A: stream K (async-staged), WMMA scores, per-lane online softmax,
//         merged across 16-lane halves with shfl_xor (wave32).
// Pass B: recompute scores, write normalized weights once to d_out,
//         LDS round-trip as bf16 A-frag, P*V via WMMA from vsT.
// ======================================================================
__global__ __launch_bounds__(256)
void attn_kernel(const __bf16* __restrict__ qh, const __bf16* __restrict__ kh,
                 const __bf16* __restrict__ vhT, const float* __restrict__ mask,
                 float* __restrict__ wts, __bf16* __restrict__ concat) {
  const int bq = blockIdx.x;
  const int h  = blockIdx.y;
  const int b  = blockIdx.z;
  const int tid = threadIdx.x, lane = tid & 31, wv = tid >> 5;

  __shared__ alignas(16) __bf16 ks[32][Dc + 8];          // K chunk [key][d]
  __shared__ alignas(16) __bf16 vsT[Dc][32 + 8];         // V chunk [d][key]
  __shared__ alignas(16) __bf16 wtile[8][16][32 + 8];    // per-wave probs [q][k]
  __shared__ float msk[Sc];

  const __bf16* qbase = qh + (((size_t)b * Hc + h) * Sc + bq * 128 + wv * 16) * Dc;
  const __bf16* kbase = kh + (((size_t)b * Hc + h) * Sc) * Dc;
  const __bf16* vtbase = vhT + (((size_t)b * Hc + h) * Dc) * Sc;

  for (int i = tid; i < Sc; i += 256) msk[i] = mask[(size_t)b * Sc + i];

  const int n_ = lane & 15;
  const int hi = lane >> 4;
  const int kb = (lane < 16) ? 0 : 8;

  // Q fragments in registers for the whole kernel (global bf16, b128 loads)
  v16bf qa[2];
#pragma unroll
  for (int half = 0; half < 2; ++half)
    qa[half] = frag_2x16B(qbase + (size_t)n_ * Dc + half * 32, kb);

  // -------- pass A --------
  float rm[8], rl[8];
#pragma unroll
  for (int r = 0; r < 8; ++r) { rm[r] = -3.0e38f; rl[r] = 0.f; }

  for (int c0 = 0; c0 < Sc; c0 += 32) {
    __syncthreads();
    if (tid < 128) {                       // 32 rows x 4 chunks of 16B
      int r = tid >> 2, q = tid & 3;
      cp16_g2l(&ks[r][q * 8], kbase + (size_t)(c0 + r) * Dc + q * 8);
    }
    if (c0 + 32 < Sc)
      __builtin_prefetch(kbase + (size_t)(c0 + 32) * Dc, 0, 0);
    async_join();
    __syncthreads();

#pragma unroll
    for (int t = 0; t < 2; ++t) {
      const __bf16* krow = &ks[t * 16 + n_][0];
      v8f sf = {};
      sf = wmma_bf16(qa[0], frag_2x16B(krow, kb), sf);
      sf = wmma_bf16(qa[1], frag_2x16B(krow + 32, kb), sf);
      const float madd = msk[c0 + t * 16 + n_] * (-1e9f);
#pragma unroll
      for (int r = 0; r < 8; ++r) {
        float s  = sf[r] * SCALE + madd;
        float mn = fmaxf(rm[r], s);
        rl[r] = rl[r] * __expf(rm[r] - mn) + __expf(s - mn);
        rm[r] = mn;
      }
    }
  }
  // merge per-row stats across the 16 lanes of each half (wave32)
#pragma unroll
  for (int r = 0; r < 8; ++r) {
    float m = rm[r], l = rl[r];
    for (int off = 1; off < 16; off <<= 1) {
      float mo = __shfl_xor(m, off, 32);
      float lo = __shfl_xor(l, off, 32);
      float mn = fmaxf(m, mo);
      l = l * __expf(m - mn) + lo * __expf(mo - mn);
      m = mn;
    }
    rm[r] = m; rl[r] = l;
  }

  // -------- pass B --------
  v8f oacc[4] = {};
  float* wrow = wts + (((size_t)b * Hc + h) * Sc + bq * 128 + wv * 16) * (size_t)Sc;

  for (int c0 = 0; c0 < Sc; c0 += 32) {
    __syncthreads();
    if (tid < 128) {
      int r = tid >> 2, q = tid & 3;
      cp16_g2l(&ks[r][q * 8], kbase + (size_t)(c0 + r) * Dc + q * 8);
    }
    {                                       // 64 rows x 4 chunks of 16B
      int d = tid >> 2;
      cp16_g2l(&vsT[d][(tid & 3) * 8], vtbase + (size_t)d * Sc + c0 + (tid & 3) * 8);
    }
    if (c0 + 32 < Sc)
      __builtin_prefetch(vtbase + c0 + 32, 0, 0);
    async_join();
    __syncthreads();

#pragma unroll
    for (int t = 0; t < 2; ++t) {
      const __bf16* krow = &ks[t * 16 + n_][0];
      v8f sf = {};
      sf = wmma_bf16(qa[0], frag_2x16B(krow, kb), sf);
      sf = wmma_bf16(qa[1], frag_2x16B(krow + 32, kb), sf);
      const float madd = msk[c0 + t * 16 + n_] * (-1e9f);
#pragma unroll
      for (int r = 0; r < 8; ++r) {
        float s = sf[r] * SCALE + madd;
        float w = __expf(s - rm[r]) / rl[r];
        const int qrow = r + hi * 8;
        wrow[(size_t)qrow * Sc + (c0 + t * 16 + n_)] = w;   // weights -> d_out
        wtile[wv][qrow][t * 16 + n_] = (__bf16)w;           // -> LDS A-frag
      }
    }
    __syncthreads();   // prob tile visibility (cross-lane within wave)

    const v16bf wa = frag_2x16B(&wtile[wv][n_][0], kb);
#pragma unroll
    for (int f = 0; f < 4; ++f) {
      const v16bf vb = frag_2x16B(&vsT[f * 16 + n_][0], kb);
      oacc[f] = wmma_bf16(wa, vb, oacc[f]);
    }
  }

  // attn -> merged-head bf16 [B,S,DM] for the output projection
#pragma unroll
  for (int f = 0; f < 4; ++f)
#pragma unroll
    for (int r = 0; r < 8; ++r) {
      const int qrow = bq * 128 + wv * 16 + r + hi * 8;
      const int d    = f * 16 + n_;
      concat[((size_t)b * Sc + qrow) * DMc + h * Dc + d] = (__bf16)oacc[f][r];
    }
}

// ======================================================================
extern "C" void kernel_launch(void* const* d_in, const int* in_sizes, int n_in,
                              void* d_out, int out_size, void* d_ws, size_t ws_size,
                              hipStream_t stream) {
  const float* Q    = (const float*)d_in[0];
  const float* K    = (const float*)d_in[1];
  const float* V    = (const float*)d_in[2];
  const float* mask = (const float*)d_in[3];
  const float* Wq_w = (const float*)d_in[4];
  const float* Wq_b = (const float*)d_in[5];
  const float* Wk_w = (const float*)d_in[6];
  const float* Wk_b = (const float*)d_in[7];
  const float* Wv_w = (const float*)d_in[8];
  const float* Wv_b = (const float*)d_in[9];
  const float* Wo_w = (const float*)d_in[10];
  const float* Wo_b = (const float*)d_in[11];

  float* out0 = (float*)d_out;                       // [B,S,DM] f32
  float* wts  = out0 + (size_t)Bc * Sc * DMc;        // [B,H,S,S] f32

  const size_t NPROJ = (size_t)Bc * Hc * Sc * Dc;    // 4.19M elems
  __bf16* qh     = (__bf16*)d_ws;                    // [B,H,S,64] bf16
  __bf16* kh     = qh + NPROJ;                       // [B,H,S,64] bf16
  __bf16* vhT    = kh + NPROJ;                       // [B,H,64,S] bf16
  __bf16* concat = vhT + NPROJ;                      // [B,S,512] bf16

  dim3 blk(256);
  dim3 g1((Bc * Sc) / 128, DMc / 64);                // (64, 8)
  gemm512<1, float><<<g1, blk, 0, stream>>>(Q, Wq_w, Wq_b, qh);
  gemm512<1, float><<<g1, blk, 0, stream>>>(K, Wk_w, Wk_b, kh);
  gemm512<2, float><<<g1, blk, 0, stream>>>(V, Wv_w, Wv_b, vhT);

  dim3 g2(Sc / 128, Hc, Bc);                         // (16, 8, 4)
  attn_kernel<<<g2, blk, 0, stream>>>(qh, kh, vhT, mask, wts, concat);

  gemm512<0, __bf16><<<g1, blk, 0, stream>>>(concat, Wo_w, Wo_b, out0);
}
